// Block_34479997452369
// MI455X (gfx1250) — compile-verified, matrix operand order checked by reading, not run on hardware
//
#include <hip/hip_runtime.h>
#include <hip/hip_bf16.h>
#include <math.h>

typedef __attribute__((ext_vector_type(16))) __bf16 v16bf;
typedef __attribute__((ext_vector_type(8)))  __bf16 v8bf;
typedef __attribute__((ext_vector_type(8)))  float  v8f;

constexpr int       kNB = 8;
constexpr int       kNT = 2048;
constexpr int       kNC = 1024;
constexpr int       kNH = 4096;
constexpr long long kBT = (long long)kNB * kNT;          // 16384 rows
constexpr long long kTotC = kBT * kNC;                   // 16.7M elems

// ---------------- fp32 -> bf16 weight/activation convert ----------------
__global__ void f32_to_bf16_kernel(const float* __restrict__ in,
                                   __bf16* __restrict__ out, long long n) {
  long long i = (long long)blockIdx.x * blockDim.x + threadIdx.x;
  long long s = (long long)gridDim.x * blockDim.x;
  for (; i < n; i += s) out[i] = (__bf16)in[i];
}

// ---------------- LayerNorm: one workgroup per row ----------------
__global__ __launch_bounds__(256) void ln_kernel(const float* __restrict__ x,
                                                 const float* __restrict__ g,
                                                 const float* __restrict__ b,
                                                 float* __restrict__ out, int C) {
  const int row = blockIdx.x;
  const float* xr = x + (size_t)row * C;
  __shared__ float s1[256], s2[256];
  float sum = 0.f, sq = 0.f;
  for (int c = threadIdx.x; c < C; c += 256) {
    float v = xr[c];
    sum += v;
    sq  += v * v;
  }
  s1[threadIdx.x] = sum; s2[threadIdx.x] = sq;
  __syncthreads();
  for (int st = 128; st > 0; st >>= 1) {
    if (threadIdx.x < st) {
      s1[threadIdx.x] += s1[threadIdx.x + st];
      s2[threadIdx.x] += s2[threadIdx.x + st];
    }
    __syncthreads();
  }
  const float m   = s1[0] / C;
  const float var = s2[0] / C - m * m;
  const float inv = rsqrtf(var + 1e-5f);
  float* orow = out + (size_t)row * C;
  for (int c = threadIdx.x; c < C; c += 256)
    orow[c] = (xr[c] - m) * inv * g[c] + b[c];
}

// ---------------- time-shift + token/channel mix, emit bf16 ----------------
__global__ void mix_kernel(const float* __restrict__ h,
                           const float* __restrict__ mk,
                           const float* __restrict__ mv,
                           const float* __restrict__ mr,
                           __bf16* __restrict__ ok,
                           __bf16* __restrict__ ov,
                           __bf16* __restrict__ orr,
                           long long total, int C, int T) {
  long long i = (long long)blockIdx.x * blockDim.x + threadIdx.x;
  long long s = (long long)gridDim.x * blockDim.x;
  for (; i < total; i += s) {
    int c = (int)(i % C);
    long long t = (i / C) % T;
    float hv = h[i];
    float hh = (t == 0) ? 0.f : h[i - C];
    float a = mk[c];
    ok[i] = (__bf16)(hv * a + hh * (1.f - a));
    if (ov) {
      float bq = mv[c];
      ov[i] = (__bf16)(hv * bq + hh * (1.f - bq));
    }
    float r = mr[c];
    orr[i] = (__bf16)(hv * r + hh * (1.f - r));
  }
}

// ---------------- bf16 WMMA GEMM: out[M,N] = A[M,K] * W[N,K]^T ----------------
// B-fragment (lane = N index, 8-wide contiguous K runs per lane) maps directly
// to row-major W[N,K]; both A and B frags are contiguous global_load_b128s.
// Each wave computes a 64x64 tile (4x4 WMMA subtiles): 16 b128 loads feed
// 16 WMMAs per K-step (~32 FLOP/byte from cache).
__device__ inline v16bf ld_frag(const __bf16* p) {
  v8bf lo = *(const v8bf*)p;         // K = base .. base+7   (per half-wave)
  v8bf hh = *(const v8bf*)(p + 16);  // K = base+16 .. base+23
  return __builtin_shufflevector(lo, hh, 0, 1, 2, 3, 4, 5, 6, 7,
                                         8, 9, 10, 11, 12, 13, 14, 15);
}

enum { GEMM_RAW = 0, GEMM_SIGMOID = 1, GEMM_RELU2_BF16 = 2, GEMM_RESADD = 3 };

__global__ __launch_bounds__(128) void gemm_bf16_kernel(
    const __bf16* __restrict__ A, const __bf16* __restrict__ W,
    float* __restrict__ outF, __bf16* __restrict__ outBf,
    const float* __restrict__ res,
    int M, int N, int K, int mode) {
  const int lane = threadIdx.x & 31;
  const int wv   = threadIdx.x >> 5;
  const int m0 = blockIdx.y * 256 + wv * 64;   // this wave's 64 rows
  const int n0 = blockIdx.x * 64;              // this block's 64 cols
  if (m0 >= M) return;

  const int l  = lane & 15;
  const int hi = (lane >> 4) & 1;

  const __bf16* abase = A + (size_t)(m0 + l) * K + hi * 8;
  const __bf16* bbase = W + (size_t)(n0 + l) * K + hi * 8;
  const size_t  astep = (size_t)16 * K;   // next 16-row subtile (imm-foldable)

  v8f acc[4][4] = {};

  for (int k0 = 0; k0 < K; k0 += 32) {
    __builtin_prefetch(abase + k0 + 64, 0, 3);   // near-scope global_prefetch_b8
    __builtin_prefetch(bbase + k0 + 64, 0, 3);
    v16bf a[4], b[4];
#pragma unroll
    for (int i = 0; i < 4; ++i) a[i] = ld_frag(abase + (size_t)i * astep + k0);
#pragma unroll
    for (int j = 0; j < 4; ++j) b[j] = ld_frag(bbase + (size_t)j * astep + k0);
#pragma unroll
    for (int i = 0; i < 4; ++i)
#pragma unroll
      for (int j = 0; j < 4; ++j)
        acc[i][j] = __builtin_amdgcn_wmma_f32_16x16x32_bf16(
            false, a[i], false, b[j], (short)0, acc[i][j], false, false);
  }

  // D layout: VGPR r, lanes 0-15 -> M=r, lanes 16-31 -> M=r+8; N = lane&15
#pragma unroll
  for (int i = 0; i < 4; ++i) {
    const int rbase = m0 + i * 16 + hi * 8;
#pragma unroll
    for (int j = 0; j < 4; ++j) {
      const int col = n0 + j * 16 + l;
#pragma unroll
      for (int r = 0; r < 8; ++r) {
        const size_t idx = (size_t)(rbase + r) * N + col;
        float v = acc[i][j][r];
        if (mode == GEMM_RAW) {
          outF[idx] = v;
        } else if (mode == GEMM_SIGMOID) {
          outF[idx] = 1.f / (1.f + expf(-v));
        } else if (mode == GEMM_RELU2_BF16) {
          float t = fmaxf(v, 0.f);
          outBf[idx] = (__bf16)(t * t);
        } else { // GEMM_RESADD
          outF[idx] = res[idx] + v;
        }
      }
    }
  }
}

// ---------------- WKV recurrence: one thread per (b, c) channel ----------------
__global__ void wkv_kernel(const float* __restrict__ td, const float* __restrict__ tf,
                           const float* __restrict__ k, const float* __restrict__ v,
                           const float* __restrict__ sr, __bf16* __restrict__ y,
                           int Bn, int Tn, int C) {
  int tid = blockIdx.x * blockDim.x + threadIdx.x;
  if (tid >= Bn * C) return;
  const int b = tid / C, c = tid % C;
  const float w = -expf(td[c]);
  const float u = tf[c];
  float aa = 0.f, bb = 0.f, pp = -1e38f;
  const long long base = (long long)b * Tn * C + c;
  for (int t = 0; t < Tn; ++t) {
    const long long idx = base + (long long)t * C;
    const float kt = k[idx], vt = v[idx];
    float ww = u + kt;
    float p  = fmaxf(pp, ww);
    float e1 = expf(pp - p), e2 = expf(ww - p);
    float yy = (e1 * aa + e2 * vt) / (e1 * bb + e2);
    y[idx] = (__bf16)(sr[idx] * yy);          // fused sigmoid(r)*wkv, bf16 for Wo GEMM
    float ww2 = pp + w;
    float p2  = fmaxf(ww2, kt);
    e1 = expf(ww2 - p2); e2 = expf(kt - p2);
    aa = e1 * aa + e2 * vt;
    bb = e1 * bb + e2;
    pp = p2;
  }
}

// ---------------- final: out = x2 + sigmoid(xr*Wcr) * kv ----------------
__global__ void final_kernel(const float* __restrict__ x2, const float* __restrict__ sr,
                             const float* __restrict__ kv, float* __restrict__ out,
                             long long n) {
  long long i = (long long)blockIdx.x * blockDim.x + threadIdx.x;
  long long s = (long long)gridDim.x * blockDim.x;
  for (; i < n; i += s) out[i] = x2[i] + sr[i] * kv[i];
}

extern "C" void kernel_launch(void* const* d_in, const int* in_sizes, int n_in,
                              void* d_out, int out_size, void* d_ws, size_t ws_size,
                              hipStream_t stream) {
  const float* x     = (const float*)d_in[0];
  const float* ln1_g = (const float*)d_in[1];
  const float* ln1_b = (const float*)d_in[2];
  const float* ln2_g = (const float*)d_in[3];
  const float* ln2_b = (const float*)d_in[4];
  const float* tdec  = (const float*)d_in[5];
  const float* tfir  = (const float*)d_in[6];
  const float* tm_k  = (const float*)d_in[7];
  const float* tm_v  = (const float*)d_in[8];
  const float* tm_r  = (const float*)d_in[9];
  const float* Wk    = (const float*)d_in[10];
  const float* Wv    = (const float*)d_in[11];
  const float* Wr    = (const float*)d_in[12];
  const float* Wo    = (const float*)d_in[13];
  const float* cm_k  = (const float*)d_in[14];
  const float* cm_r  = (const float*)d_in[15];
  const float* Wck   = (const float*)d_in[16];
  const float* Wcr   = (const float*)d_in[17];
  const float* Wcv   = (const float*)d_in[18];
  float* out = (float*)d_out;

  // ---- workspace carve-up (256B aligned slots) ----
  char* wp = (char*)d_ws;
  auto take = [&](size_t bytes) {
    char* p = wp;
    wp += (bytes + 255) & ~(size_t)255;
    return p;
  };
  const size_t CC = (size_t)kNC * kNC, CH = (size_t)kNC * kNH;
  __bf16* wkb  = (__bf16*)take(CC * 2);
  __bf16* wvb  = (__bf16*)take(CC * 2);
  __bf16* wrb  = (__bf16*)take(CC * 2);
  __bf16* wob  = (__bf16*)take(CC * 2);
  __bf16* wckb = (__bf16*)take(CH * 2);
  __bf16* wcrb = (__bf16*)take(CC * 2);
  __bf16* wcvb = (__bf16*)take(CH * 2);
  float*  h    = (float*) take(kTotC * 4);        // LN output (reused for ln2)
  __bf16* xk   = (__bf16*)take(kTotC * 2);        // reused as xk2
  __bf16* xv   = (__bf16*)take(kTotC * 2);
  __bf16* xr   = (__bf16*)take(kTotC * 2);        // reused as xr2
  float*  kbuf = (float*) take(kTotC * 4);        // k; reused as kv
  float*  vbuf = (float*) take(kTotC * 4);        // v; reused as sr2
  float*  rbuf = (float*) take(kTotC * 4);        // sigmoid(r)
  __bf16* rwkv = (__bf16*)take(kTotC * 2);
  float*  x2   = (float*) take(kTotC * 4);
  __bf16* kk   = (__bf16*)take(kBT * (size_t)kNH * 2);
  (void)ws_size; (void)n_in; (void)in_sizes; (void)out_size;

  const dim3 blk128(128), blk256(256);
  const dim3 gCxC(kNC / 64, (unsigned)(kBT / 256));   // 64-wide N, 256-tall M blocks
  const dim3 gHxC(kNH / 64, (unsigned)(kBT / 256));

  // weights -> bf16
  f32_to_bf16_kernel<<<256, blk256, 0, stream>>>(Wk,  wkb,  (long long)CC);
  f32_to_bf16_kernel<<<256, blk256, 0, stream>>>(Wv,  wvb,  (long long)CC);
  f32_to_bf16_kernel<<<256, blk256, 0, stream>>>(Wr,  wrb,  (long long)CC);
  f32_to_bf16_kernel<<<256, blk256, 0, stream>>>(Wo,  wob,  (long long)CC);
  f32_to_bf16_kernel<<<512, blk256, 0, stream>>>(Wck, wckb, (long long)CH);
  f32_to_bf16_kernel<<<256, blk256, 0, stream>>>(Wcr, wcrb, (long long)CC);
  f32_to_bf16_kernel<<<512, blk256, 0, stream>>>(Wcv, wcvb, (long long)CH);

  // ---- time-mix branch ----
  ln_kernel<<<(unsigned)kBT, blk256, 0, stream>>>(x, ln1_g, ln1_b, h, kNC);
  mix_kernel<<<8192, blk256, 0, stream>>>(h, tm_k, tm_v, tm_r, xk, xv, xr,
                                          kTotC, kNC, kNT);
  gemm_bf16_kernel<<<gCxC, blk128, 0, stream>>>(xk, wkb, kbuf, nullptr, nullptr,
                                                (int)kBT, kNC, kNC, GEMM_RAW);
  gemm_bf16_kernel<<<gCxC, blk128, 0, stream>>>(xv, wvb, vbuf, nullptr, nullptr,
                                                (int)kBT, kNC, kNC, GEMM_RAW);
  gemm_bf16_kernel<<<gCxC, blk128, 0, stream>>>(xr, wrb, rbuf, nullptr, nullptr,
                                                (int)kBT, kNC, kNC, GEMM_SIGMOID);
  wkv_kernel<<<(kNB * kNC + 255) / 256, blk256, 0, stream>>>(
      tdec, tfir, kbuf, vbuf, rbuf, rwkv, kNB, kNT, kNC);
  gemm_bf16_kernel<<<gCxC, blk128, 0, stream>>>(rwkv, wob, x2, nullptr, x,
                                                (int)kBT, kNC, kNC, GEMM_RESADD);

  // ---- channel-mix branch ----
  ln_kernel<<<(unsigned)kBT, blk256, 0, stream>>>(x2, ln2_g, ln2_b, h, kNC);
  mix_kernel<<<8192, blk256, 0, stream>>>(h, cm_k, nullptr, cm_r, xk, nullptr, xr,
                                          kTotC, kNC, kNT);
  gemm_bf16_kernel<<<gHxC, blk128, 0, stream>>>(xk, wckb, nullptr, kk, nullptr,
                                                (int)kBT, kNH, kNC, GEMM_RELU2_BF16);
  gemm_bf16_kernel<<<gCxC, blk128, 0, stream>>>(kk, wcvb, kbuf /*kv*/, nullptr, nullptr,
                                                (int)kBT, kNC, kNH, GEMM_RAW);
  gemm_bf16_kernel<<<gCxC, blk128, 0, stream>>>(xr, wcrb, vbuf /*sr2*/, nullptr, nullptr,
                                                (int)kBT, kNC, kNC, GEMM_SIGMOID);
  final_kernel<<<8192, blk256, 0, stream>>>(x2, vbuf, kbuf, out, kTotC);
}
